// OpticalConv2d_82600811037162
// MI455X (gfx1250) — matrix operand order repaired and checked by reading, not compiled, not run on hardware
//
#include <hip/hip_runtime.h>

// ---------------------------------------------------------------------------
// OpticalConv2d on MI455X (gfx1250). Reference FFT path == circular conv:
//   out[b,o,u,v] = bias[o] + sum_{c,i,j} K[o,c,i,j] * img[b,c,(u+4-i)%116,(v+4-j)%116]
// (img = x zero-padded by 2 to 116x116). ~1060 FLOP/byte -> compute-bound, so
// implicit GEMM on v_wmma_f32_16x16x32_bf16 with a 2-term bf16 split
// (hi*hi + hi*lo + lo*hi) for fp32-class accuracy at bf16-WMMA rate.
//
// Key layout trick: v16bf word r' holds K-elements (2r',2r'+1) for both A and
// B fragments, so LDS / permuted weights store bf16 pairs of CONSECUTIVE
// channels packed in one u32, hi-plane and lo-plane separate. Fragments are
// then loaded directly into WMMA register layout with 2x b128 each — no
// per-tap unpack VALU at all.
// ---------------------------------------------------------------------------

typedef __attribute__((ext_vector_type(16))) __bf16 v16bf;
typedef __attribute__((ext_vector_type(8)))  float v8f;

union Frag { uint4 q[2]; v16bf v; };   // 8 u32 = one 16x16x32 bf16 operand

#define BEAM 116
#define SPX  112
#define CIN  64
#define COUT 128
#define NB   16
// Slab: 10 rows x 22 cols; per (r,d): words 0..31 = hi pairs, 32..63 = lo
// pairs, 64..67 pad (stride 68 => 16B aligned, 4-bank stride between lanes).
#define SLAB_R 10
#define SLAB_D 22
#define SLAB_CSTRIDE 68

__device__ inline unsigned short f2bf(float f) {
  unsigned int u = __float_as_uint(f);
  return (unsigned short)((u + 0x7FFFu + ((u >> 16) & 1u)) >> 16); // RNE
}
__device__ inline float bf2f(unsigned short h) {
  return __uint_as_float(((unsigned int)h) << 16);
}
__device__ inline unsigned int pack2(unsigned short a, unsigned short b) {
  return (unsigned int)a | ((unsigned int)b << 16);
}

// ---------------------------------------------------------------------------
// Kernel 1: permute weights (COUT,CIN,7,7) f32 into per-lane WMMA B-fragment
// words, bf16 pairs, hi-plane then lo-plane per lane:
//   Wt[(((t*49+tap)*2+kbi)*32+lane)*16 + hl*8 + r']
// B 32x16 bf16: lanes 0-15 K=e, lanes 16-31 K=16+e; N=lane%16. Word r' packs
// K=(2r',2r'+1). 1.6 MB total, stays hot in L2.
// ---------------------------------------------------------------------------
__global__ __launch_bounds__(256)
void oc_wt_transform(const float* __restrict__ W, unsigned int* __restrict__ Wt) {
  int idx = blockIdx.x * 256 + threadIdx.x;
  const int total = 8 * 49 * 2 * 32 * 16;
  if (idx >= total) return;
  int rp   = idx & 7;
  int hl   = (idx >> 3) & 1;
  int lane = (idx >> 4) & 31;
  int kbi  = (idx >> 9) & 1;
  int tapT = idx >> 10;
  int tap  = tapT % 49;
  int t    = tapT / 49;
  int o    = t * 16 + (lane & 15);
  int c0   = kbi * 32 + ((lane < 16) ? 0 : 16) + 2 * rp;
  float w0 = W[(o * CIN + c0) * 49 + tap];
  float w1 = W[(o * CIN + c0 + 1) * 49 + tap];
  unsigned short h0 = f2bf(w0), h1 = f2bf(w1);
  if (hl == 0) {
    Wt[idx] = pack2(h0, h1);
  } else {
    Wt[idx] = pack2(f2bf(w0 - bf2f(h0)), f2bf(w1 - bf2f(h1)));
  }
}

// ---------------------------------------------------------------------------
// Kernel 2: main conv. Workgroup = (b, 4 u-rows, 16 v-cols); 8 waves, one
// 16-cout tile each. Per tap iteration: 24 WMMA, 32 ds_b128, 8 global_b128.
// ---------------------------------------------------------------------------
__global__ __launch_bounds__(256)
void oc_conv_main(const float* __restrict__ X, const unsigned int* __restrict__ Wt,
                  const float* __restrict__ bias, float* __restrict__ out) {
  __shared__ unsigned int S[SLAB_R * SLAB_D * SLAB_CSTRIDE]; // ~59.8 KB

  const int v0 = blockIdx.x * 16;   // 8 tiles (last ragged: 116 = 7*16+4)
  const int u0 = blockIdx.y * 4;    // 29 blocks
  const int b  = blockIdx.z;
  const int tid = threadIdx.x;

  // ---- stage input slab: hi-plane + lo-plane, channel pairs packed -------
  // S[r][d][w]    = bf16hi(c=2w) | bf16hi(c=2w+1)<<16
  // S[r][d][32+w] = bf16lo pair.  img index math handles wrap + zero-pad.
  for (int idx = tid; idx < SLAB_R * SLAB_D * 32; idx += 256) {
    int w  = idx & 31;
    int rd = idx >> 5;
    int d  = rd % SLAB_D;
    int r  = rd / SLAB_D;
    int q  = (u0 - 2 + r + BEAM) % BEAM;
    int p  = (v0 - 2 + d + BEAM) % BEAM;
    float v0f = 0.0f, v1f = 0.0f;
    if (q >= 2 && q <= 113 && p >= 2 && p <= 113) {
      const float* src = X + ((b * CIN + 2 * w) * SPX + (q - 2)) * SPX + (p - 2);
      v0f = src[0];
      v1f = src[SPX * SPX];
    }
    unsigned short h0 = f2bf(v0f), h1 = f2bf(v1f);
    unsigned int* slot = &S[(r * SLAB_D + d) * SLAB_CSTRIDE + w];
    slot[0]  = pack2(h0, h1);
    slot[32] = pack2(f2bf(v0f - bf2f(h0)), f2bf(v1f - bf2f(h1)));
  }
  __syncthreads();

  const int wave = tid >> 5;      // cout tile
  const int lane = tid & 31;
  const int m    = lane & 15;     // M = v offset (A rows)
  const int half = lane >> 4;
  const int h4   = half * 8;      // A-layout: upper half-wave reads K+16 run

  v8f acc[4] = {v8f{}, v8f{}, v8f{}, v8f{}};

#pragma unroll 1
  for (int tap = 0; tap < 49; ++tap) {
    const int i = tap / 7, j = tap % 7;
    const int d = m + 6 - j;            // slab col for output v = v0+m
#pragma unroll
    for (int kbi = 0; kbi < 2; ++kbi) {
      const int kb2 = kbi * 16;         // word offset of this 32-channel chunk

      // ---- B fragment: 4 contiguous b128, pre-formatted hi|lo ----------
      const uint4* bp = (const uint4*)(Wt +
          ((((wave * 49 + tap) * 2 + kbi) * 32 + lane) << 4));
      Frag bh, bl;
      bh.q[0] = bp[0]; bh.q[1] = bp[1];
      bl.q[0] = bp[2]; bl.q[1] = bp[3];

#pragma unroll
      for (int uu = 0; uu < 4; ++uu) {
        const int r = uu + 6 - i;       // slab row for output u = u0+uu
        const unsigned int* base = &S[(r * SLAB_D + d) * SLAB_CSTRIDE];
        Frag ah, al;
        // lanes 0-15: K {0..15}; lanes 16-31: K {16..31}
        ah.q[0] = *(const uint4*)(base + kb2 + h4);
        ah.q[1] = *(const uint4*)(base + kb2 + h4 + 4);
        al.q[0] = *(const uint4*)(base + 32 + kb2 + h4);
        al.q[1] = *(const uint4*)(base + 32 + kb2 + h4 + 4);

        // 2-term split: hi*hi + hi*lo + lo*hi (lo*lo below fp32 ulp)
        acc[uu] = __builtin_amdgcn_wmma_f32_16x16x32_bf16(
            false, ah.v, false, bh.v, (short)0, acc[uu], false, false);
        acc[uu] = __builtin_amdgcn_wmma_f32_16x16x32_bf16(
            false, ah.v, false, bl.v, (short)0, acc[uu], false, false);
        acc[uu] = __builtin_amdgcn_wmma_f32_16x16x32_bf16(
            false, al.v, false, bh.v, (short)0, acc[uu], false, false);
      }
    }
  }

  // ---- epilogue: C/D layout lane%16 = N (cout), VGPR r = M (+8 upper) ----
  const int o  = wave * 16 + m;
  const float bv = bias[o];
#pragma unroll
  for (int uu = 0; uu < 4; ++uu) {
    const int u = u0 + uu;
#pragma unroll
    for (int r = 0; r < 8; ++r) {
      const int v = v0 + r + half * 8;
      if (v < BEAM)
        out[((b * COUT + o) * BEAM + u) * BEAM + v] = acc[uu][r] + bv;
    }
  }
}

extern "C" void kernel_launch(void* const* d_in, const int* in_sizes, int n_in,
                              void* d_out, int out_size, void* d_ws, size_t ws_size,
                              hipStream_t stream) {
  (void)in_sizes; (void)n_in; (void)out_size; (void)ws_size;
  const float* x    = (const float*)d_in[0];   // (16,64,112,112)
  const float* w    = (const float*)d_in[1];   // (128,64,7,7)
  const float* bias = (const float*)d_in[2];   // (128,1,1)
  float* out        = (float*)d_out;           // (16,128,116,116)
  unsigned int* Wt  = (unsigned int*)d_ws;     // 401408 u32 = 1.6 MB

  const int nw = 8 * 49 * 2 * 32 * 16;
  oc_wt_transform<<<(nw + 255) / 256, 256, 0, stream>>>(w, Wt);

  dim3 grid(8, 29, NB);   // v-tiles x u-blocks x batch
  oc_conv_main<<<grid, 256, 0, stream>>>(x, Wt, bias, out);
}
